// Encoder_45054206935436
// MI455X (gfx1250) — matrix operand order repaired and checked by reading, not compiled, not run on hardware
//
#include <hip/hip_runtime.h>
#include <hip/hip_bf16.h>

// ---------------------------------------------------------------------------
// GRU encoder for MI455X (gfx1250, wave32, WMMA + TDM async weight staging).
// V=32000 E=1024 U=1024 B=64 S=512
// ---------------------------------------------------------------------------

#define V_SZ 32000
#define E_SZ 1024
#define U_SZ 1024
#define B_SZ 64
#define S_SZ 512
#define N3   3072            // 3*U

typedef __bf16 v16bf __attribute__((ext_vector_type(16)));
typedef __bf16 v8bf  __attribute__((ext_vector_type(8)));
typedef float  v8f   __attribute__((ext_vector_type(8)));
typedef unsigned short v8us __attribute__((ext_vector_type(8)));
typedef unsigned int   v4u  __attribute__((ext_vector_type(4)));
typedef int            v4i  __attribute__((ext_vector_type(4)));
typedef int            v8i  __attribute__((ext_vector_type(8)));

union BF16x16 { v16bf v; v8bf h[2]; };

__device__ __forceinline__ unsigned short f2bf(float f) {
  union { float f; unsigned u; } v; v.f = f;
  unsigned r = v.u + 0x7FFFu + ((v.u >> 16) & 1u);   // round-to-nearest-even
  return (unsigned short)(r >> 16);
}
__device__ __forceinline__ float bf2f(unsigned short h) {
  union { unsigned u; float f; } v; v.u = ((unsigned)h) << 16; return v.f;
}
__device__ __forceinline__ v8f vzero() {
  v8f z;
  #pragma unroll
  for (int i = 0; i < 8; ++i) z[i] = 0.0f;
  return z;
}
__device__ __forceinline__ v8f wmma_bf16(v16bf a, v16bf b, v8f c) {
  // D = A(16x32 bf16) * B(32x16 bf16) + C(16x16 f32)
  return __builtin_amdgcn_wmma_f32_16x16x32_bf16(false, a, false, b,
                                                 (short)0, c, false, false);
}
__device__ __forceinline__ float sigmoidf(float x) {
  return 1.0f / (1.0f + __expf(-x));
}

// A-fragment: 16x32 bf16 tile, row-major source with ld elements.
// lane l: row M = m0 + (l&15); K base kb = (l<16)?0:16; regs = K kb..kb+15 (contiguous)
__device__ __forceinline__ v16bf load_a_frag(const unsigned short* A, int ld,
                                             int m0, int k0) {
  int l  = threadIdx.x & 31;
  int kb = (l < 16) ? 0 : 16;
  const unsigned short* p = A + (size_t)(m0 + (l & 15)) * ld + k0 + kb;
  BF16x16 u;
  u.h[0] = *(const v8bf*)(const void*)(p);
  u.h[1] = *(const v8bf*)(const void*)(p + 8);
  return u.v;
}
// B-fragment: 32x16 bf16 tile from N-major (transposed) weights, ld elements per N row.
// lane l: col N = n0 + (l&15); K half kb = (l<16)?0:16; regs = K kb..kb+15 (contiguous)
__device__ __forceinline__ v16bf load_b_frag(const unsigned short* Wt, int ld,
                                             int n0, int k0) {
  int l  = threadIdx.x & 31;
  int kb = (l < 16) ? 0 : 16;
  return *(const v16bf*)(const void*)(Wt + (size_t)(n0 + (l & 15)) * ld + k0 + kb);
}

// Issue a TDM 2D tile load: rows x 32 bf16 elements from an N-major weight
// matrix (row stride `srcld` elements) into contiguous LDS at lds_byte_addr.
// D# layout per CDNA5 ISA ch.8.3-8.5.
__device__ __forceinline__ void tdm_load_tile(const unsigned short* src_tile,
                                              unsigned lds_byte_addr,
                                              unsigned rows, unsigned srcld,
                                              unsigned tens_rows) {
  unsigned long long ga = (unsigned long long)(const void*)src_tile;
  v4u g0;
  g0[0] = 1u;                                   // count=1, is_restore=0
  g0[1] = lds_byte_addr;                        // lds_addr
  g0[2] = (unsigned)ga;                         // global_addr[31:0]
  g0[3] = (unsigned)((ga >> 32) & 0x01FFFFFFu)  // global_addr[56:32]
          | 0x80000000u;                        // type=2 ("image")
  v8i g1;
  g1[0] = (int)(1u << 16);                      // data_size=1 (2 bytes)
  g1[1] = (int)((srcld & 0xFFFFu) << 16);       // tensor_dim0[15:0] @ bit48
  g1[2] = (int)(((srcld >> 16) & 0xFFFFu) |     // tensor_dim0[31:16]
                ((tens_rows & 0xFFFFu) << 16)); // tensor_dim1[15:0]
  g1[3] = (int)(((tens_rows >> 16) & 0xFFFFu) | // tensor_dim1[31:16]
                (32u << 16));                   // tile_dim0 = 32
  g1[4] = (int)rows;                            // tile_dim1 = rows, tile_dim2=0
  g1[5] = (int)srcld;                           // tensor_dim0_stride[31:0]
  g1[6] = 0;                                    // stride0[47:32], stride1[15:0]
  g1[7] = 0;
  v4i gz; gz[0] = gz[1] = gz[2] = gz[3] = 0;
#if __has_include(<hip/amd_detail/amd_gfx1250_TDM.h>)
  v8i gz8;
  #pragma unroll
  for (int i = 0; i < 8; ++i) gz8[i] = 0;
  __builtin_amdgcn_tensor_load_to_lds(g0, g1, gz, gz, gz8, 0);
#else
  __builtin_amdgcn_tensor_load_to_lds(g0, g1, gz, gz, 0);
#endif
}

// ---------------------------------------------------------------------------
__global__ void k_init(unsigned* bar) {
  if (threadIdx.x < 8) bar[threadIdx.x] = 0u;
}

// cast + transpose: kT[n][e] = kernel[e][n], rT[n][u] = rkernel[u][n]  (bf16)
__global__ void k_prep(const float* __restrict__ ker, const float* __restrict__ rker,
                       unsigned short* __restrict__ kT, unsigned short* __restrict__ rT) {
  int tid = blockIdx.x * blockDim.x + threadIdx.x;
  const int half = N3 * E_SZ;                 // 3072*1024
  if (tid < half) {
    int e = tid & (E_SZ - 1), n = tid >> 10;
    kT[tid] = f2bf(ker[(size_t)e * N3 + n]);
  } else {
    int o = tid - half;
    int u = o & (U_SZ - 1), n = o >> 10;
    rT[o] = f2bf(rker[(size_t)u * N3 + n]);
  }
}

// embedding gather -> bf16 x[B*S][E]
__global__ void k_gather(const int* __restrict__ wid, const float* __restrict__ emb,
                         unsigned short* __restrict__ xb) {
  int tid = blockIdx.x * blockDim.x + threadIdx.x;   // [0, 32768*128)
  int m = tid >> 7;
  int e0 = (tid & 127) << 3;
  const float4* p = (const float4*)(emb + (size_t)wid[m] * E_SZ + e0);
  float4 f0 = p[0], f1 = p[1];
  v8us o;
  o[0] = f2bf(f0.x); o[1] = f2bf(f0.y); o[2] = f2bf(f0.z); o[3] = f2bf(f0.w);
  o[4] = f2bf(f1.x); o[5] = f2bf(f1.y); o[6] = f2bf(f1.z); o[7] = f2bf(f1.w);
  *(v8us*)(xb + (size_t)m * E_SZ + e0) = o;
}

// xp[s][b][n] = x[b*S+s][:] . kernel[:][n] + bias0[n]   (bf16 out, time-major)
// WG: 256 thr = 8 waves stacked in M. WG tile 128(M) x 128(N).
// Weight tiles (128 x 32 bf16, 8 KB) DMA'd into LDS by the TDM, double-buffered:
// wave0 issues tile k+1 before the WMMA work on tile k, waits TENSORcnt at the end.
__global__ void __launch_bounds__(256)
k_gemm_xp(const unsigned short* __restrict__ xb, const unsigned short* __restrict__ kT,
          const float* __restrict__ bias, unsigned short* __restrict__ xp) {
  __shared__ __align__(128) unsigned short sB[2][128 * 32];   // 2 x 8 KB
  const int t = threadIdx.x, wave = t >> 5, lane = t & 31;
  const int m0 = blockIdx.x * 128 + wave * 16;
  const int n0 = blockIdx.y * 128;

  v8f acc[8];
  #pragma unroll
  for (int i = 0; i < 8; ++i) acc[i] = vzero();

  // prologue: DMA tile k0=0 into buffer 0
  if (wave == 0) {
    tdm_load_tile(kT + (size_t)n0 * E_SZ,
                  (unsigned)(size_t)(&sB[0][0]), 128, E_SZ, N3);
    __builtin_amdgcn_s_wait_tensorcnt(0);
  }
  __syncthreads();

  int p = 0;
  for (int k0 = 0; k0 < E_SZ; k0 += 32) {
    if (wave == 0 && (k0 + 32) < E_SZ) {    // overlap: DMA next tile
      tdm_load_tile(kT + (size_t)n0 * E_SZ + (k0 + 32),
                    (unsigned)(size_t)(&sB[p ^ 1][0]), 128, E_SZ, N3);
    }
    __builtin_prefetch(xb + (size_t)(m0 + (lane & 15)) * E_SZ + k0 + 64, 0, 1);
    v16bf a = load_a_frag(xb, E_SZ, m0, k0);
    #pragma unroll
    for (int strip = 0; strip < 8; ++strip) {
      v16bf b = load_b_frag(&sB[p][0], 32, strip * 16, 0);
      acc[strip] = wmma_bf16(a, b, acc[strip]);
    }
    if (wave == 0) __builtin_amdgcn_s_wait_tensorcnt(0);
    __syncthreads();
    p ^= 1;
  }
  // epilogue: +bias0, cast, scatter time-major
  #pragma unroll
  for (int strip = 0; strip < 8; ++strip) {
    int n = n0 + strip * 16 + (lane & 15);
    float b0 = bias[n];
    #pragma unroll
    for (int r = 0; r < 8; ++r) {
      int m = m0 + r + ((lane < 16) ? 0 : 8);
      int bb = m >> 9, ss = m & (S_SZ - 1);
      xp[((size_t)ss * B_SZ + bb) * N3 + n] = f2bf(acc[strip][r] + b0);
    }
  }
}

// ---------------------------------------------------------------------------
__device__ __forceinline__ void grid_sync(unsigned* cnt, volatile unsigned* gen,
                                          unsigned nwg) {
  __threadfence();
  __syncthreads();
  if (threadIdx.x == 0) {
    unsigned g = *gen;
    if (atomicAdd(cnt, 1u) == nwg - 1u) {
      *cnt = 0u;
      __threadfence();
      atomicAdd((unsigned*)gen, 1u);
    } else {
      while (*gen == g) __builtin_amdgcn_s_sleep(2);
    }
  }
  __syncthreads();
}

// Persistent GRU scan. 48 WGs x 128 thr (4 waves). 192 waves -> 192 N-strips of 16.
// Each step: rec = h_bf16 @ Rt + bias1 (WMMA), gsync, gates (f32), gsync.
__global__ void __launch_bounds__(128)
k_scan(const unsigned short* __restrict__ xp, const unsigned short* __restrict__ rT,
       const float* __restrict__ bias, const unsigned char* __restrict__ mask,
       float* __restrict__ rec, float* __restrict__ hf,
       unsigned short* __restrict__ hb, float* __restrict__ out,
       unsigned* __restrict__ bar) {
  const int t = threadIdx.x, wave = t >> 5, lane = t & 31;
  const int gw = blockIdx.x * 4 + wave;          // 0..191
  const int n0 = gw * 16;
  const int bn = n0 + (lane & 15);               // this lane's output column
  const float b1 = bias[N3 + bn];
  unsigned* cnt = bar;
  volatile unsigned* gen = bar + 1;
  const size_t BSU = (size_t)B_SZ * S_SZ * U_SZ;

  // h := 0
  if (blockIdx.x < 32) {
    #pragma unroll
    for (int i = 0; i < 16; ++i) {
      int e = i * 128 + t;                       // 0..2047
      int b = e >> 5, j = blockIdx.x * 32 + (e & 31);
      hf[b * U_SZ + j] = 0.0f;
      hb[b * U_SZ + j] = 0;
    }
  }
  grid_sync(cnt, gen, 48);

  for (int s = 0; s < S_SZ; ++s) {
    // ---- rec = h @ R + bias1, this wave's 16 columns, all 64 batch rows ----
    v8f a0 = vzero(), a1 = vzero(), a2 = vzero(), a3 = vzero();
    const unsigned short* rrow =
        rT + (size_t)bn * U_SZ + ((lane < 16) ? 0 : 16);
    for (int k0 = 0; k0 < U_SZ; k0 += 32) {
      __builtin_prefetch(rrow + k0 + 128, 0, 1);
      v16bf bfr = *(const v16bf*)(const void*)(rrow + k0);
      v16bf f0 = load_a_frag(hb, U_SZ, 0,  k0);
      v16bf f1 = load_a_frag(hb, U_SZ, 16, k0);
      v16bf f2 = load_a_frag(hb, U_SZ, 32, k0);
      v16bf f3 = load_a_frag(hb, U_SZ, 48, k0);
      a0 = wmma_bf16(f0, bfr, a0);
      a1 = wmma_bf16(f1, bfr, a1);
      a2 = wmma_bf16(f2, bfr, a2);
      a3 = wmma_bf16(f3, bfr, a3);
    }
    #pragma unroll
    for (int mt = 0; mt < 4; ++mt) {
      v8f a = (mt == 0) ? a0 : (mt == 1) ? a1 : (mt == 2) ? a2 : a3;
      #pragma unroll
      for (int r = 0; r < 8; ++r) {
        int m = mt * 16 + r + ((lane < 16) ? 0 : 8);
        rec[m * N3 + bn] = a[r] + b1;
      }
    }
    grid_sync(cnt, gen, 48);

    // ---- gates: h' = z*h + (1-z)*tanh(xh + r*rh), masked carry ----
    if (blockIdx.x < 32) {
      int j0 = blockIdx.x * 32;
      #pragma unroll 4
      for (int i = 0; i < 16; ++i) {
        int e = i * 128 + t;
        int b = e >> 5, j = j0 + (e & 31);
        size_t xo = ((size_t)s * B_SZ + b) * N3 + j;
        float xz = bf2f(xp[xo]);
        float xr = bf2f(xp[xo + U_SZ]);
        float xh = bf2f(xp[xo + 2 * U_SZ]);
        float rz = rec[b * N3 + j];
        float rr = rec[b * N3 + j + U_SZ];
        float rh = rec[b * N3 + j + 2 * U_SZ];
        float h  = hf[b * U_SZ + j];
        float z  = sigmoidf(xz + rz);
        float r  = sigmoidf(xr + rr);
        float hh = tanhf(xh + r * rh);
        float hn = z * h + (1.0f - z) * hh;
        if (!mask[b * S_SZ + s]) hn = h;
        hf[b * U_SZ + j] = hn;
        hb[b * U_SZ + j] = f2bf(hn);
        out[((size_t)b * S_SZ + s) * U_SZ + j] = hn;
        if (s == S_SZ - 1) out[BSU + (size_t)b * U_SZ + j] = hn;
      }
    }
    grid_sync(cnt, gen, 48);
  }
}

// ---------------------------------------------------------------------------
extern "C" void kernel_launch(void* const* d_in, const int* in_sizes, int n_in,
                              void* d_out, int out_size, void* d_ws, size_t ws_size,
                              hipStream_t stream) {
  (void)in_sizes; (void)n_in; (void)out_size; (void)ws_size;
  const int*           word_ids = (const int*)d_in[0];
  const unsigned char* mask     = (const unsigned char*)d_in[1];
  const float*         emb      = (const float*)d_in[2];
  const float*         ker      = (const float*)d_in[3];
  const float*         rker     = (const float*)d_in[4];
  const float*         bias     = (const float*)d_in[5];
  float*               out      = (float*)d_out;

  char* ws = (char*)d_ws;
  size_t off = 0;
  unsigned*       bar = (unsigned*)(ws + off);        off += 256;
  unsigned short* kT  = (unsigned short*)(ws + off);  off += (size_t)N3 * E_SZ * 2;
  unsigned short* rT  = (unsigned short*)(ws + off);  off += (size_t)N3 * U_SZ * 2;
  unsigned short* xb  = (unsigned short*)(ws + off);  off += (size_t)B_SZ * S_SZ * E_SZ * 2;
  unsigned short* xp  = (unsigned short*)(ws + off);  off += (size_t)B_SZ * S_SZ * N3 * 2;
  float*          rec = (float*)(ws + off);           off += (size_t)B_SZ * N3 * 4;
  float*          hf  = (float*)(ws + off);           off += (size_t)B_SZ * U_SZ * 4;
  unsigned short* hb  = (unsigned short*)(ws + off);  off += (size_t)B_SZ * U_SZ * 2;

  k_init<<<1, 64, 0, stream>>>(bar);
  k_prep<<<(2 * N3 * E_SZ) / 256, 256, 0, stream>>>(ker, rker, kT, rT);
  k_gather<<<(B_SZ * S_SZ * (E_SZ / 8)) / 256, 256, 0, stream>>>(word_ids, emb, xb);
  dim3 g1(B_SZ * S_SZ / 128, N3 / 128);   // 256 x 24
  k_gemm_xp<<<g1, 256, 0, stream>>>(xb, kT, bias, xp);
  k_scan<<<48, 128, 0, stream>>>(xp, rT, bias, mask, rec, hf, hb, out, bar);
}
